// MobilityBERTMoE_14285061227050
// MI455X (gfx1250) — matrix operand order repaired and checked by reading, not compiled, not run on hardware
//
#include <hip/hip_runtime.h>
#include <math.h>

// ---- model dims ----
#define B_    16
#define HIST_ 336
#define PRED_ 48
#define S_    384
#define T_    (B_*S_)      // 6144 tokens
#define H_    768
#define FF_   3072
#define V_    40000
#define NE_   8
#define DIN_  481
#define DPAD_ 512

typedef __attribute__((ext_vector_type(16))) _Float16 v16h;
typedef __attribute__((ext_vector_type(8)))  float    v8f;
typedef __attribute__((ext_vector_type(4)))  float    v4f;

// load 8 consecutive fp32 (16B-aligned, global or LDS) and convert into
// fragment slots [base..base+7]  (emits b128 loads + v_cvt_pk_f16_f32)
__device__ __forceinline__ void cvt8_to(const float* p, v16h& a, int base) {
  v4f x = *(const v4f*)p;
  v4f y = *(const v4f*)(p + 4);
  a[base+0] = (_Float16)x[0]; a[base+1] = (_Float16)x[1];
  a[base+2] = (_Float16)x[2]; a[base+3] = (_Float16)x[3];
  a[base+4] = (_Float16)y[0]; a[base+5] = (_Float16)y[1];
  a[base+6] = (_Float16)y[2]; a[base+7] = (_Float16)y[3];
}

// CDNA5 async global->LDS copies (ASYNCcnt-tracked; LDS addr in VDST VGPR,
// 64-bit global addr in VADDR pair; see CDNA5 ISA)
__device__ __forceinline__ void async_copy_b128(void* lds, const float* g) {
  unsigned la = (unsigned)(size_t)lds;           // low 32 bits = LDS byte offset
  asm volatile("global_load_async_to_lds_b128 %0, %1, off"
               :: "v"(la), "v"((unsigned long long)(size_t)g) : "memory");
}
__device__ __forceinline__ void async_copy_b32(void* lds, const float* g) {
  unsigned la = (unsigned)(size_t)lds;
  asm volatile("global_load_async_to_lds_b32 %0, %1, off"
               :: "v"(la), "v"((unsigned long long)(size_t)g) : "memory");
}
__device__ __forceinline__ void wait_async0() {
  asm volatile("s_wait_asynccnt 0x0" ::: "memory");
}

// stage one 64x32 A tile (row-major) and one 32x128 W tile (transposed to
// n-major) into LDS buffers; async fast path for fully in-bounds tiles,
// guarded zero-filling synchronous fallback otherwise.  No waits here —
// caller overlaps these copies with compute on the other buffer.
__device__ __forceinline__ void stage_tiles(float (*Asf)[36], float (*Bsf)[36],
                                            const float* __restrict__ A, int lda,
                                            const float* __restrict__ W,
                                            int M, int N, int K,
                                            int bM, int bN, int k0, int tid)
{
  bool full = (bM + 64 <= M) && (bN + 128 <= N) && (k0 + 32 <= K);
  if (full) {
    // A tile: two b128 async copies per thread (row-major -> row-major)
    #pragma unroll
    for (int i = 0; i < 2; ++i) {
      int idx = tid + i*256;
      int r = idx >> 3, c4 = (idx & 7) * 4;
      async_copy_b128(&Asf[r][c4], &A[(size_t)(bM + r)*lda + k0 + c4]);
    }
    // W tile: 16 b32 async copies per thread; coalesced global reads,
    // per-lane LDS scatter performs the n-major transpose
    #pragma unroll
    for (int i = 0; i < 16; ++i) {
      int idx = tid + i*256;
      int kl = idx >> 7, nl = idx & 127;
      async_copy_b32(&Bsf[nl][kl], &W[(size_t)(k0 + kl)*N + bN + nl]);
    }
  } else {
    #pragma unroll
    for (int i = 0; i < 8; ++i) {
      int idx = tid + i*256;
      int r = idx >> 5, cc = idx & 31;
      int m = bM + r, k = k0 + cc;
      Asf[r][cc] = (m < M && k < K) ? A[(size_t)m*lda + k] : 0.f;
    }
    #pragma unroll
    for (int i = 0; i < 16; ++i) {
      int idx = tid + i*256;
      int kl = idx >> 7, nl = idx & 127;
      int k = k0 + kl, n = bN + nl;
      Bsf[nl][kl] = (k < K && n < N) ? W[(size_t)k*N + n] : 0.f;
    }
  }
}

// ===================== embedding gather + input LayerNorm =====================
__global__ void embed_ln_kernel(const int* __restrict__ isf, const int* __restrict__ hloc,
                                const int* __restrict__ psf,
                                const float* __restrict__ day_e, const float* __restrict__ time_e,
                                const float* __restrict__ dow_e, const float* __restrict__ wd_e,
                                const float* __restrict__ loc_e,
                                const float* __restrict__ g, const float* __restrict__ b,
                                float* __restrict__ xpad)
{
  int lane = threadIdx.x & 31;
  int wave = threadIdx.x >> 5;
  int t = blockIdx.x * 8 + wave;
  if (t >= T_) return;
  int bb = t / S_, s = t % S_;
  int f0, f1, f3, f4; float dur; int loci = -1;
  if (s < HIST_) {
    const int* f = isf + (size_t)(bb*HIST_ + s) * 6;
    f0 = f[0]; f1 = f[1]; f3 = f[3]; f4 = f[4]; dur = (float)f[5];
    loci = hloc[bb*HIST_ + s];
  } else {
    const int* f = psf + (size_t)(bb*PRED_ + (s - HIST_)) * 6;
    f0 = f[0]; f1 = f[1]; f3 = f[3]; f4 = f[4]; dur = (float)f[5];
  }
  float vals[16];
  float sum = 0.f;
  #pragma unroll
  for (int it = 0; it < 16; ++it) {
    int i = lane + it * 32;
    float v = 0.f;
    if (i < DIN_) {
      if      (i < 64)  v = day_e [f0*64 + i];
      else if (i < 128) v = time_e[f1*64 + (i-64)];
      else if (i < 192) v = dow_e [f3*64 + (i-128)];
      else if (i < 224) v = wd_e  [f4*32 + (i-192)];
      else if (i == 224) v = dur;
      else              v = (loci >= 0) ? loc_e[(size_t)loci*256 + (i-225)] : 0.f;
    }
    vals[it] = v; sum += v;
  }
  #pragma unroll
  for (int m = 16; m >= 1; m >>= 1) sum += __shfl_xor(sum, m, 32);
  float mean = sum / (float)DIN_;
  float vs = 0.f;
  #pragma unroll
  for (int it = 0; it < 16; ++it) {
    int i = lane + it * 32;
    if (i < DIN_) { float d = vals[it] - mean; vs += d*d; }
  }
  #pragma unroll
  for (int m = 16; m >= 1; m >>= 1) vs += __shfl_xor(vs, m, 32);
  float inv = rsqrtf(vs / (float)DIN_ + 1e-5f);
  float* row = xpad + (size_t)t * DPAD_;
  #pragma unroll
  for (int it = 0; it < 16; ++it) {
    int i = lane + it * 32;
    float o = 0.f;
    if (i < DIN_) o = (vals[it] - mean) * inv * g[i] + b[i];
    row[i] = o;                           // covers 0..511; pad lanes write 0
  }
}

// ===================== LayerNorm over H=768 (wave per row) =====================
__global__ void ln_kernel(const float* __restrict__ X, const float* __restrict__ R,
                          const float* __restrict__ pos, const float* __restrict__ tt,
                          const float* __restrict__ g, const float* __restrict__ bta,
                          float eps, float* __restrict__ out)
{
  int lane = threadIdx.x & 31;
  int wave = threadIdx.x >> 5;
  int t = blockIdx.x * 8 + wave;
  if (t >= T_) return;
  int s = t % S_;
  float vals[24]; float sum = 0.f;
  #pragma unroll
  for (int it = 0; it < 24; ++it) {
    int i = lane + it * 32;
    float v = X[(size_t)t*H_ + i];
    if (R)   v += R[(size_t)t*H_ + i];
    if (pos) v += pos[(size_t)s*H_ + i];
    if (tt)  v += tt[i];
    vals[it] = v; sum += v;
  }
  #pragma unroll
  for (int m = 16; m >= 1; m >>= 1) sum += __shfl_xor(sum, m, 32);
  float mean = sum * (1.f / (float)H_);
  float vs = 0.f;
  #pragma unroll
  for (int it = 0; it < 24; ++it) { float d = vals[it] - mean; vs += d*d; }
  #pragma unroll
  for (int m = 16; m >= 1; m >>= 1) vs += __shfl_xor(vs, m, 32);
  float inv = rsqrtf(vs * (1.f / (float)H_) + eps);
  #pragma unroll
  for (int it = 0; it < 24; ++it) {
    int i = lane + it * 32;
    out[(size_t)t*H_ + i] = (vals[it] - mean) * inv * g[i] + bta[i];
  }
}

// ===================== generic WMMA GEMM (double-buffered async) ==============
// C[M,N] = epilogue( A[M,K](lda) @ W[K,N] + bias[N] )
// 256 threads = 8 waves (2x4); block tile 64(M) x 128(N); wave tile 32x32
// (2x2 fragments -> 4 WMMAs/K-step with A/B reuse).
// Two LDS buffers: async copies for tile t+1 are issued before computing on
// tile t, then a single s_wait_asynccnt 0 + barrier per iteration; copy
// latency hides under the WMMAs.  LDS row stride = 36 floats (144B) keeps
// fragment reads aligned ds_load_b128, converted to f16 at fragment build.
// ACT: 0=none, 1=exact gelu.  MOE: scale by gate[row*8+expert], optional C+=.
template<int ACT, bool MOE>
__global__ void __launch_bounds__(256)
gemm_wmma(const float* __restrict__ A, int lda,
          const float* __restrict__ W,
          const float* __restrict__ bias,
          float* __restrict__ C, int ldc,
          int M, int N, int K,
          const float* __restrict__ gate, int expert, int accum)
{
  __shared__ float Asf[2][64][36];
  __shared__ float Bsf[2][128][36];
  int tid  = threadIdx.x;
  int lane = tid & 31, wave = tid >> 5;
  int wm = wave >> 2, wn = wave & 3;           // 2 x 4 wave grid
  int row = lane & 15, half = lane >> 4;
  int bM = blockIdx.y * 64, bN = blockIdx.x * 128;
  int nsteps = (K + 31) / 32;
  v8f acc[2][2] = {};

  stage_tiles(Asf[0], Bsf[0], A, lda, W, M, N, K, bM, bN, 0, tid);
  wait_async0();
  __syncthreads();

  int buf = 0;
  for (int t = 0; t < nsteps; ++t) {
    int k0 = t * 32;
    // issue copies for the next tile into the other buffer (no wait)
    if (t + 1 < nsteps)
      stage_tiles(Asf[buf^1], Bsf[buf^1], A, lda, W, M, N, K, bM, bN, k0 + 32, tid);
    // prefetch the tile after next into cache (global_prefetch_b8); L2 is 192MB
    if (k0 + 64 < K) {
      int pm = bM + (tid & 63);
      if (pm < M) __builtin_prefetch(&A[(size_t)pm*lda + k0 + 64], 0, 1);
      int pk = k0 + 64 + (tid >> 7), pn = bN + (tid & 127);
      if (pk < K && pn < N) __builtin_prefetch(&W[(size_t)pk*N + pn], 0, 1);
    }
    // compute on the current buffer (overlaps with async copies)
    v16h afr[2], bfr[2];
    #pragma unroll
    for (int mi = 0; mi < 2; ++mi) {
      int ar = wm*32 + mi*16 + row;
      cvt8_to(&Asf[buf][ar][16*half],     afr[mi], 0);
      cvt8_to(&Asf[buf][ar][16*half + 8], afr[mi], 8);
    }
    #pragma unroll
    for (int ni = 0; ni < 2; ++ni) {
      int bc = wn*32 + ni*16 + row;
      cvt8_to(&Bsf[buf][bc][16*half],     bfr[ni], 0);
      cvt8_to(&Bsf[buf][bc][16*half + 8], bfr[ni], 8);
    }
    #pragma unroll
    for (int mi = 0; mi < 2; ++mi)
      #pragma unroll
      for (int ni = 0; ni < 2; ++ni)
        acc[mi][ni] = __builtin_amdgcn_wmma_f32_16x16x32_f16(
            false, afr[mi], false, bfr[ni], (short)0, acc[mi][ni], false, false);
    // one wait + one barrier per iteration: after this, every wave has
    // finished reading buf and finished its copies into buf^1
    wait_async0();
    __syncthreads();
    buf ^= 1;
  }

  #pragma unroll
  for (int mi = 0; mi < 2; ++mi) {
    #pragma unroll
    for (int ni = 0; ni < 2; ++ni) {
      int col = bN + wn*32 + ni*16 + row;
      #pragma unroll
      for (int j = 0; j < 8; ++j) {
        int rm = bM + wm*32 + mi*16 + half*8 + j;
        if (rm < M && col < N) {
          float vv = acc[mi][ni][j];
          if (bias) vv += bias[col];
          if (ACT == 1) vv = 0.5f * vv * (1.f + erff(vv * 0.70710678118654752f));
          if (MOE) {
            vv *= gate[(size_t)rm*NE_ + expert];
            if (accum) vv += C[(size_t)rm*ldc + col];
          }
          C[(size_t)rm*ldc + col] = vv;
        }
      }
    }
  }
}

// ===================== head split / merge =====================
// trans==0: out [b,h,S,64] (pad d>=hd with 0);  trans==1: out [b,h,64,S] (V^T)
__global__ void split_heads(const float* __restrict__ in, float* __restrict__ out,
                            int nh, int hd, int trans)
{
  size_t idx = (size_t)blockIdx.x * blockDim.x + threadIdx.x;
  size_t total = (size_t)B_ * nh * S_ * 64;
  if (idx >= total) return;
  int d, s; size_t bh;
  if (!trans) { d = (int)(idx & 63); s = (int)((idx >> 6) % S_); bh = (idx >> 6) / S_; }
  else        { s = (int)(idx % S_); size_t r = idx / S_; d = (int)(r & 63); bh = r >> 6; }
  int head = (int)(bh % nh); int bb = (int)(bh / nh);
  float v = 0.f;
  if (d < hd) v = in[(size_t)(bb*S_ + s)*H_ + head*hd + d];
  out[idx] = v;
}

__global__ void merge_heads(const float* __restrict__ ctxp, float* __restrict__ out,
                            int nh, int hd)
{
  size_t idx = (size_t)blockIdx.x * blockDim.x + threadIdx.x;
  if (idx >= (size_t)T_ * H_) return;
  int cch = (int)(idx % H_); size_t t = idx / H_;
  int head = cch / hd, d = cch % hd;
  int bb = (int)(t / S_), s = (int)(t % S_);
  out[idx] = ctxp[((size_t)(bb*nh + head)*S_ + s)*64 + d];
}

// ===================== attention: scores = scale * Q K^T =====================
// one wave per 16x16 score tile; Qp/Kp: [bh][S][64] (d padded to 64 with 0).
// K row-major [k][d] is exactly the n-major/k-contiguous B layout; all
// fragment loads are aligned float4 (global_load_b128) + cvt to f16.
__global__ void __launch_bounds__(32)
attn_scores(const float* __restrict__ Qp, const float* __restrict__ Kp,
            float* __restrict__ scores, float scale)
{
  int lane = threadIdx.x;
  int row = lane & 15, half = lane >> 4;
  int kt = blockIdx.x, qt = blockIdx.y, bh = blockIdx.z;
  const float* Q  = Qp + (size_t)bh * S_ * 64;
  const float* Kb = Kp + (size_t)bh * S_ * 64;
  v8f c = {};
  #pragma unroll
  for (int step = 0; step < 2; ++step) {
    int k0 = step * 32;
    v16h a, b;
    const float* qp = Q  + (size_t)(qt*16 + row)*64 + k0;
    const float* kp = Kb + (size_t)(kt*16 + row)*64 + k0 + 16*half;
    cvt8_to(qp + 16*half,     a, 0);
    cvt8_to(qp + 16*half + 8, a, 8);
    cvt8_to(kp,               b, 0);
    cvt8_to(kp + 8,           b, 8);
    c = __builtin_amdgcn_wmma_f32_16x16x32_f16(false, a, false, b, (short)0, c, false, false);
  }
  float* out = scores + (size_t)bh * S_ * S_;
  #pragma unroll
  for (int j = 0; j < 8; ++j) {
    int q = qt*16 + half*8 + j;
    int k = kt*16 + row;
    out[(size_t)q*S_ + k] = c[j] * scale;
  }
}

// ===================== softmax over rows of length 384 (wave per row) =========
__global__ void softmax_rows(float* __restrict__ p, int rows)
{
  int lane = threadIdx.x & 31, wave = threadIdx.x >> 5;
  int r = blockIdx.x * 8 + wave;
  if (r >= rows) return;
  float* row = p + (size_t)r * S_;
  float v[12]; float mx = -1e30f;
  #pragma unroll
  for (int i = 0; i < 12; ++i) { v[i] = row[lane + i*32]; mx = fmaxf(mx, v[i]); }
  #pragma unroll
  for (int m = 16; m >= 1; m >>= 1) mx = fmaxf(mx, __shfl_xor(mx, m, 32));
  float sum = 0.f;
  #pragma unroll
  for (int i = 0; i < 12; ++i) { v[i] = expf(v[i] - mx); sum += v[i]; }
  #pragma unroll
  for (int m = 16; m >= 1; m >>= 1) sum += __shfl_xor(sum, m, 32);
  float inv = 1.f / sum;
  #pragma unroll
  for (int i = 0; i < 12; ++i) row[lane + i*32] = v[i] * inv;
}

// ===================== attention: ctx = P @ V =====================
// Vt: [bh][64][S]  (transposed so K pairs are contiguous for the B fragment)
__global__ void __launch_bounds__(32)
attn_pv(const float* __restrict__ probs, const float* __restrict__ Vt,
        float* __restrict__ ctxp)
{
  int lane = threadIdx.x;
  int row = lane & 15, half = lane >> 4;
  int dt = blockIdx.x, qt = blockIdx.y, bh = blockIdx.z;
  const float* P  = probs + (size_t)bh * S_ * S_;
  const float* Vb = Vt    + (size_t)bh * 64 * S_;
  v8f c = {};
  for (int k0 = 0; k0 < S_; k0 += 32) {
    v16h a, b;
    const float* pp = P  + (size_t)(qt*16 + row)*S_ + k0;
    const float* vp = Vb + (size_t)(dt*16 + row)*S_ + k0 + 16*half;
    cvt8_to(pp + 16*half,     a, 0);
    cvt8_to(pp + 16*half + 8, a, 8);
    cvt8_to(vp,               b, 0);
    cvt8_to(vp + 8,           b, 8);
    c = __builtin_amdgcn_wmma_f32_16x16x32_f16(false, a, false, b, (short)0, c, false, false);
  }
  float* out = ctxp + (size_t)bh * S_ * 64;
  #pragma unroll
  for (int j = 0; j < 8; ++j) {
    int q = qt*16 + half*8 + j;
    int d = dt*16 + row;
    out[(size_t)q*64 + d] = c[j];
  }
}

// ===================== MoE gate: softmax(h @ gate_W + gate_b) ================
__global__ void moe_gate(const float* __restrict__ h, const float* __restrict__ gW,
                         const float* __restrict__ gb, float* __restrict__ gp)
{
  int t = blockIdx.x * blockDim.x + threadIdx.x;
  if (t >= T_) return;
  float lg[NE_];
  #pragma unroll
  for (int e = 0; e < NE_; ++e) lg[e] = gb[e];
  const float* hr = h + (size_t)t * H_;
  for (int d = 0; d < H_; ++d) {
    float hv = hr[d];
    #pragma unroll
    for (int e = 0; e < NE_; ++e) lg[e] += hv * gW[d*NE_ + e];
  }
  float mx = lg[0];
  #pragma unroll
  for (int e = 1; e < NE_; ++e) mx = fmaxf(mx, lg[e]);
  float sum = 0.f;
  #pragma unroll
  for (int e = 0; e < NE_; ++e) { lg[e] = expf(lg[e] - mx); sum += lg[e]; }
  float inv = 1.f / sum;
  #pragma unroll
  for (int e = 0; e < NE_; ++e) gp[(size_t)t*NE_ + e] = lg[e] * inv;
}

// ===================== gather prediction rows: moe + res =====================
__global__ void gather_pred(const float* __restrict__ moe, const float* __restrict__ res,
                            float* __restrict__ out)
{
  size_t idx = (size_t)blockIdx.x * blockDim.x + threadIdx.x;
  if (idx >= (size_t)(B_*PRED_) * H_) return;
  int m = (int)(idx / H_), cch = (int)(idx % H_);
  int bb = m / PRED_, r = m % PRED_;
  size_t t = (size_t)bb * S_ + HIST_ + r;
  out[idx] = moe[t*H_ + cch] + res[t*H_ + cch];
}

// ============================ host launch =====================================
extern "C" void kernel_launch(void* const* d_in, const int* in_sizes, int n_in,
                              void* d_out, int out_size, void* d_ws, size_t ws_size,
                              hipStream_t stream)
{
  (void)in_sizes; (void)n_in; (void)out_size; (void)ws_size;
  const int*   isf    = (const int*)  d_in[0];
  const int*   hloc   = (const int*)  d_in[1];
  const int*   psf    = (const int*)  d_in[2];
  const float* day_e  = (const float*)d_in[3];
  const float* time_e = (const float*)d_in[4];
  const float* dow_e  = (const float*)d_in[5];
  const float* wd_e   = (const float*)d_in[6];
  const float* loc_e  = (const float*)d_in[7];
  const float* lng    = (const float*)d_in[8];
  const float* lnb    = (const float*)d_in[9];
  const float* inp_W  = (const float*)d_in[10];
  const float* inp_b  = (const float*)d_in[11];
  const float* res_W  = (const float*)d_in[12];
  const float* res_b  = (const float*)d_in[13];
  const float* sa_qW  = (const float*)d_in[14];
  const float* sa_kW  = (const float*)d_in[15];
  const float* sa_vW  = (const float*)d_in[16];
  const float* sa_oW  = (const float*)d_in[17];
  const float* sa_qb  = (const float*)d_in[18];
  const float* sa_kb  = (const float*)d_in[19];
  const float* sa_vb  = (const float*)d_in[20];
  const float* sa_ob  = (const float*)d_in[21];
  const float* sa_lng = (const float*)d_in[22];
  const float* sa_lnb = (const float*)d_in[23];
  const float* pos_e  = (const float*)d_in[24];
  const float* tt_e   = (const float*)d_in[25];
  const float* be_lng = (const float*)d_in[26];
  const float* be_lnb = (const float*)d_in[27];
  const float* bqW    = (const float*)d_in[28];
  const float* bkW    = (const float*)d_in[29];
  const float* bvW    = (const float*)d_in[30];
  const float* boW    = (const float*)d_in[31];
  const float* bqb    = (const float*)d_in[32];
  const float* bkb    = (const float*)d_in[33];
  const float* bvb    = (const float*)d_in[34];
  const float* bob    = (const float*)d_in[35];
  const float* balng  = (const float*)d_in[36];
  const float* balnb  = (const float*)d_in[37];
  const float* bf1W   = (const float*)d_in[38];
  const float* bf1b   = (const float*)d_in[39];
  const float* bf2W   = (const float*)d_in[40];
  const float* bf2b   = (const float*)d_in[41];
  const float* bflng  = (const float*)d_in[42];
  const float* bflnb  = (const float*)d_in[43];
  const float* moe_W  = (const float*)d_in[44];
  const float* moe_b  = (const float*)d_in[45];
  const float* gate_W = (const float*)d_in[46];
  const float* gate_b = (const float*)d_in[47];
  const float* out_W  = (const float*)d_in[48];
  const float* out_b  = (const float*)d_in[49];

  float* ws = (float*)d_ws;
  size_t off = 0;
  auto alloc = [&](size_t n) { float* p = ws + off; off += n; return p; };
  float* xpad  = alloc((size_t)T_ * DPAD_);
  float* proj  = alloc((size_t)T_ * H_);
  float* resb  = alloc((size_t)T_ * H_);
  float* hbuf  = alloc((size_t)T_ * H_);
  float* abuf  = alloc((size_t)T_ * H_);
  float* qbuf  = alloc((size_t)T_ * H_);
  float* kbuf  = alloc((size_t)T_ * H_);
  float* vbuf  = alloc((size_t)T_ * H_);
  float* ctxH  = alloc((size_t)T_ * H_);
  float* attno = alloc((size_t)T_ * H_);
  float* ffbuf = alloc((size_t)T_ * FF_);
  float* Qp    = alloc((size_t)B_ * 16 * S_ * 64);
  float* Kp    = alloc((size_t)B_ * 16 * S_ * 64);
  float* Vt    = alloc((size_t)B_ * 16 * S_ * 64);
  float* ctxp  = alloc((size_t)B_ * 16 * S_ * 64);
  float* probs = alloc((size_t)B_ * 16 * S_ * S_);
  float* moea  = alloc((size_t)T_ * H_);
  float* gpb   = alloc((size_t)T_ * NE_);
  float* predin= alloc((size_t)(B_*PRED_) * H_);

  auto gemm = [&](const float* A, int lda, const float* W, const float* bias,
                  float* C, int ldc, int M, int N, int K, int act) {
    dim3 g((unsigned)((N + 127) / 128), (unsigned)((M + 63) / 64));
    if (act)
      gemm_wmma<1, false><<<g, 256, 0, stream>>>(A, lda, W, bias, C, ldc, M, N, K,
                                                 nullptr, 0, 0);
    else
      gemm_wmma<0, false><<<g, 256, 0, stream>>>(A, lda, W, bias, C, ldc, M, N, K,
                                                 nullptr, 0, 0);
  };

  // 1) embeddings + input LN -> xpad [T,512]
  embed_ln_kernel<<<T_/8, 256, 0, stream>>>(isf, hloc, psf, day_e, time_e, dow_e,
                                            wd_e, loc_e, lng, lnb, xpad);
  // 2) input + residual projections (K=481, A row stride 512)
  gemm(xpad, DPAD_, inp_W, inp_b, proj, H_, T_, H_, DIN_, 0);
  gemm(xpad, DPAD_, res_W, res_b, resb, H_, T_, H_, DIN_, 0);

  auto attention = [&](const float* x, int nh, int hd,
                       const float* qW, const float* qb, const float* kW, const float* kb,
                       const float* vW, const float* vb) {
    gemm(x, H_, qW, qb, qbuf, H_, T_, H_, H_, 0);
    gemm(x, H_, kW, kb, kbuf, H_, T_, H_, H_, 0);
    gemm(x, H_, vW, vb, vbuf, H_, T_, H_, H_, 0);
    size_t tot = (size_t)B_ * nh * S_ * 64;
    int blocks = (int)((tot + 255) / 256);
    split_heads<<<blocks, 256, 0, stream>>>(qbuf, Qp, nh, hd, 0);
    split_heads<<<blocks, 256, 0, stream>>>(kbuf, Kp, nh, hd, 0);
    split_heads<<<blocks, 256, 0, stream>>>(vbuf, Vt, nh, hd, 1);
    dim3 gs(S_/16, S_/16, (unsigned)(B_*nh));
    attn_scores<<<gs, 32, 0, stream>>>(Qp, Kp, probs, 1.0f / sqrtf((float)hd));
    int rows = B_ * nh * S_;
    softmax_rows<<<(rows + 7) / 8, 256, 0, stream>>>(probs, rows);
    dim3 gv(4, S_/16, (unsigned)(B_*nh));
    attn_pv<<<gv, 32, 0, stream>>>(probs, Vt, ctxp);
    merge_heads<<<(int)(((size_t)T_*H_ + 255)/256), 256, 0, stream>>>(ctxp, ctxH, nh, hd);
  };

  // 3) extra 16-head self-attention block with post-LN residual (eps 1e-5)
  attention(proj, 16, 48, sa_qW, sa_qb, sa_kW, sa_kb, sa_vW, sa_vb);
  gemm(ctxH, H_, sa_oW, sa_ob, attno, H_, T_, H_, H_, 0);
  ln_kernel<<<T_/8, 256, 0, stream>>>(attno, proj, nullptr, nullptr,
                                      sa_lng, sa_lnb, 1e-5f, hbuf);
  // 4) BERT embeddings: LN(h + pos + tt), eps 1e-12
  ln_kernel<<<T_/8, 256, 0, stream>>>(hbuf, nullptr, pos_e, tt_e,
                                      be_lng, be_lnb, 1e-12f, hbuf);
  // 5) 12 BERT encoder layers (post-LN)
  for (int i = 0; i < 12; ++i) {
    const float* qW = bqW + (size_t)i*H_*H_;   const float* qb2 = bqb + (size_t)i*H_;
    const float* kW = bkW + (size_t)i*H_*H_;   const float* kb2 = bkb + (size_t)i*H_;
    const float* vW = bvW + (size_t)i*H_*H_;   const float* vb2 = bvb + (size_t)i*H_;
    const float* oW = boW + (size_t)i*H_*H_;   const float* ob2 = bob + (size_t)i*H_;
    attention(hbuf, 12, 64, qW, qb2, kW, kb2, vW, vb2);
    gemm(ctxH, H_, oW, ob2, attno, H_, T_, H_, H_, 0);
    ln_kernel<<<T_/8, 256, 0, stream>>>(attno, hbuf, nullptr, nullptr,
                                        balng + (size_t)i*H_, balnb + (size_t)i*H_,
                                        1e-12f, abuf);
    gemm(abuf, H_, bf1W + (size_t)i*H_*FF_, bf1b + (size_t)i*FF_,
         ffbuf, FF_, T_, FF_, H_, 1 /*gelu*/);
    gemm(ffbuf, FF_, bf2W + (size_t)i*FF_*H_, bf2b + (size_t)i*H_,
         attno, H_, T_, H_, FF_, 0);
    ln_kernel<<<T_/8, 256, 0, stream>>>(attno, abuf, nullptr, nullptr,
                                        bflng + (size_t)i*H_, bflnb + (size_t)i*H_,
                                        1e-12f, hbuf);
  }
  // 6) MoE: gate probs, then gate-scaled expert accumulation
  moe_gate<<<(T_ + 255)/256, 256, 0, stream>>>(hbuf, gate_W, gate_b, gpb);
  for (int e = 0; e < NE_; ++e) {
    dim3 g((H_ + 127)/128, (T_ + 63)/64);
    gemm_wmma<1, true><<<g, 256, 0, stream>>>(hbuf, H_, moe_W + (size_t)e*H_*H_,
                                              moe_b + (size_t)e*H_, moea, H_,
                                              T_, H_, H_, gpb, e, e > 0 ? 1 : 0);
  }
  // 7) gather predicted rows (moe + res) and project to vocab
  gather_pred<<<(int)(((size_t)(B_*PRED_)*H_ + 255)/256), 256, 0, stream>>>(moea, resb, predin);
  gemm(predin, H_, out_W, out_b, (float*)d_out, V_, B_*PRED_, V_, H_, 0);
}
